// GeoFeat_28046136443283
// MI455X (gfx1250) — matrix operand order repaired and checked by reading, not compiled, not run on hardware
//
#include <hip/hip_runtime.h>
#include <hip/hip_bf16.h>
#include <math.h>

typedef __bf16 bf16_t;
typedef __attribute__((ext_vector_type(16))) __bf16 bf16x16;
typedef __attribute__((ext_vector_type(8)))  __bf16 bf16x8;
typedef __attribute__((ext_vector_type(8)))  float  f32x8;

__device__ __forceinline__ bf16_t f2bf(float f) { return (bf16_t)f; }

// ---------------------------------------------------------------------------
// Elementwise helpers
// ---------------------------------------------------------------------------
__global__ void zero_f32(float* __restrict__ p, int count) {
    int i = blockIdx.x * blockDim.x + threadIdx.x;
    if (i < count) p[i] = 0.0f;
}

__global__ void cast_f32_to_bf16(const float* __restrict__ in,
                                 bf16_t* __restrict__ out, size_t count) {
    size_t stride = (size_t)gridDim.x * blockDim.x;
    for (size_t i = (size_t)blockIdx.x * blockDim.x + threadIdx.x; i < count; i += stride)
        out[i] = f2bf(in[i]);
}

// BN finalize: scale = g * rsqrt(var + eps), shift = be - mean*scale
__global__ void bn_finalize(const float* __restrict__ colsum,
                            const float* __restrict__ colsumsq,
                            const float* __restrict__ g,
                            const float* __restrict__ be,
                            float invE,
                            float* __restrict__ scale,
                            float* __restrict__ shift) {
    int n = threadIdx.x;  // 128 threads
    float m   = colsum[n] * invE;
    float var = colsumsq[n] * invE - m * m;
    float sc  = g[n] * rsqrtf(var + 1e-5f);
    scale[n] = sc;
    shift[n] = be[n] - m * sc;
}

// y = relu(x*scale + shift) -> bf16 ; X is rows x 128
__global__ void bn_relu_cast(const float* __restrict__ X,
                             const float* __restrict__ scale,
                             const float* __restrict__ shift,
                             bf16_t* __restrict__ out, size_t count) {
    size_t stride = (size_t)gridDim.x * blockDim.x;
    for (size_t i = (size_t)blockIdx.x * blockDim.x + threadIdx.x; i < count; i += stride) {
        int n = (int)(i & 127);
        float v = fmaf(X[i], scale[n], shift[n]);
        out[i] = f2bf(fmaxf(v, 0.0f));
    }
}

// ---------------------------------------------------------------------------
// Weight packing into WMMA B-fragment order (bf16, 16x16x32 shape).
// B layout (32x16 bf16): lane L holds column n = ntile*16 + (L&15);
// lane half h holds k_local 8h..8h+7 (j<8) and 16+8h..16+8h+7 (j>=8),
// packed contiguously (16 bf16 = 32 bytes per lane per k-tile).
// packed[((ntile*KT + kt)*32 + lane)*16 + j] = W[k, n], k = kt*32+(j&8)*2+(L>>4)*8+(j&7)
// ---------------------------------------------------------------------------
__global__ void pack_weights(const float* __restrict__ W, int N, int Kvalid,
                             int KT, int ntiles, bf16_t* __restrict__ out) {
    int idx = blockIdx.x * blockDim.x + threadIdx.x;
    int total = ntiles * KT * 32 * 16;
    if (idx >= total) return;
    int j    = idx & 15;
    int lane = (idx >> 4) & 31;
    int kt   = (idx >> 9) % KT;
    int nt   = idx / (KT * 512);
    int n = nt * 16 + (lane & 15);
    int k = kt * 32 + (j & 8) * 2 + (lane >> 4) * 8 + (j & 7);
    float v = (k < Kvalid) ? W[(size_t)k * N + n] : 0.0f;
    out[idx] = f2bf(v);
}

// ---------------------------------------------------------------------------
// Wave-tiled WMMA GEMM:  Out[M x (ntiles*16)] = A[M x (KT*32)] @ Bpacked + bias
// Each wave owns one 16-column N-tile, holds all B fragments in VGPRs,
// loops over M-tiles with a double-buffered A-fragment pipeline.
// Requires M % 16 == 0 (true for all launches here). LDO is the output
// leading dimension (template const so store offsets fold to immediates).
// ---------------------------------------------------------------------------
template <int KT, int LDO, bool COLSTATS, bool ACCUM, bool STOREF, bool STOREB>
__global__ __launch_bounds__(256) void gemm_wmma(
    const bf16_t* __restrict__ A, int lda,
    const bf16_t* __restrict__ Bp,
    const float* __restrict__ bias,
    float* __restrict__ Of, bf16_t* __restrict__ Ob,
    int M, int ntiles,
    float* __restrict__ colsum, float* __restrict__ colsumsq,
    int mtilesPerBlock) {
    const int wave = threadIdx.x >> 5;
    const int lane = threadIdx.x & 31;
    if (wave >= ntiles) return;
    const int laneN  = lane & 15;
    const int laneHi = lane >> 4;
    const int n = wave * 16 + laneN;

    // Load this wave's B fragments (contiguous 32B per lane per k-tile).
    bf16x16 bfrag[KT];
    const bf16x16* bv = (const bf16x16*)Bp;
#pragma unroll
    for (int kt = 0; kt < KT; ++kt)
        bfrag[kt] = bv[((size_t)wave * KT + kt) * 32 + lane];

    const float bias_n = bias ? bias[n] : 0.0f;
    float s = 0.0f, sq = 0.0f;

    const int mbase0 = blockIdx.x * mtilesPerBlock * 16;
    if (mbase0 >= M) return;
    int ntile_iters = (M - mbase0) / 16;
    if (ntile_iters > mtilesPerBlock) ntile_iters = mtilesPerBlock;

    // Incremental pointers (no per-tile 64-bit muls).
    const bf16_t* aptr = A + (size_t)(mbase0 + laneN) * lda + laneHi * 8;
    const size_t oof = (size_t)(mbase0 + laneHi * 8) * LDO + n;
    float*  ofp = (STOREF || ACCUM) ? (Of + oof) : nullptr;
    bf16_t* obp = STOREB ? (Ob + oof) : nullptr;
    const size_t astep = (size_t)16 * lda;

    union frag { bf16x16 v; bf16x8 h[2]; };

    for (int t = 0; t < ntile_iters; ++t) {
        // Prefetch next M-tile's A row.
        __builtin_prefetch(aptr + astep, 0, 1);

        f32x8 acc;
#pragma unroll
        for (int r = 0; r < 8; ++r) acc[r] = bias_n;

        // Double-buffered A-fragment pipeline: issue loads for k-tile kt+1
        // before the WMMA consuming k-tile kt, so loads overlap the matrix op.
        frag a_cur, a_nxt;
        a_cur.h[0] = *(const bf16x8*)(aptr + 0);
        a_cur.h[1] = *(const bf16x8*)(aptr + 16);
#pragma unroll
        for (int kt = 0; kt < KT; ++kt) {
            if (kt + 1 < KT) {
                a_nxt.h[0] = *(const bf16x8*)(aptr + (kt + 1) * 32);
                a_nxt.h[1] = *(const bf16x8*)(aptr + (kt + 1) * 32 + 16);
            }
            acc = __builtin_amdgcn_wmma_f32_16x16x32_bf16(
                false, a_cur.v, false, bfrag[kt], (short)0, acc, false, false);
            a_cur = a_nxt;
        }

        // Straight-line epilogue: immediate store offsets (r*LDO folds).
#pragma unroll
        for (int r = 0; r < 8; ++r) {
            float v = acc[r];
            if (ACCUM)  v += ofp[r * LDO];
            if (STOREF) ofp[r * LDO] = v;
            if (STOREB) obp[r * LDO] = f2bf(v);
            if (COLSTATS) { s += v; sq += v * v; }
        }
        aptr += astep;
        if (STOREF || ACCUM) ofp += (size_t)16 * LDO;
        if (STOREB) obp += (size_t)16 * LDO;
    }
    if (COLSTATS) {
        atomicAdd(&colsum[n], s);
        atomicAdd(&colsumsq[n], sq);
    }
}

// ---------------------------------------------------------------------------
// Per-edge geometry: one wave per edge (lane n = vector index 0..31).
// Builds the 441-wide (padded to 448) bf16 feature row H[e].
// ---------------------------------------------------------------------------
__global__ __launch_bounds__(256) void build_H(
    const float* __restrict__ Vloc,   // N x 96
    const float* __restrict__ Vedge,  // E x 96
    const float* __restrict__ rot,    // E x 9
    const float* __restrict__ trans,  // E x 3
    const int* __restrict__ eidx,     // 2 x E
    int E, bf16_t* __restrict__ H) {
    int lane = threadIdx.x & 31;
    int e = (int)(((size_t)blockIdx.x * blockDim.x + threadIdx.x) >> 5);
    if (e >= E) return;
    int src = eidx[e];
    int dst = eidx[E + e];

    const float* R = rot + (size_t)e * 9;
    float r0 = R[0], r1 = R[1], r2 = R[2];
    float r3 = R[3], r4 = R[4], r5 = R[5];
    float r6 = R[6], r7 = R[7], r8 = R[8];
    float t0 = trans[(size_t)e * 3 + 0];
    float t1 = trans[(size_t)e * 3 + 1];
    float t2 = trans[(size_t)e * 3 + 2];

    const float* ve = Vedge + (size_t)e * 96 + lane * 3;
    const float* vs = Vloc + (size_t)src * 96 + lane * 3;
    const float* vd = Vloc + (size_t)dst * 96 + lane * 3;
    float ve0 = ve[0], ve1 = ve[1], ve2 = ve[2];
    float vs0 = vs[0], vs1 = vs[1], vs2 = vs[2];
    float vd0 = vd[0], vd1 = vd[1], vd2 = vd[2];

    // Qt for Ks[lane] = V_edge[lane]
    float qe0 = r0 * ve0 + r1 * ve1 + r2 * ve2 + t0;
    float qe1 = r3 * ve0 + r4 * ve1 + r5 * ve2 + t1;
    float qe2 = r6 * ve0 + r7 * ve1 + r8 * ve2 + t2;
    // RKs = R @ Vs ; Qt for Ks[lane+32] = Vs[lane]
    float rk0 = r0 * vs0 + r1 * vs1 + r2 * vs2;
    float rk1 = r3 * vs0 + r4 * vs1 + r5 * vs2;
    float rk2 = r6 * vs0 + r7 * vs1 + r8 * vs2;
    float qs0 = rk0 + t0, qs1 = rk1 + t1, qs2 = rk2 + t2;
    float qrk = vd0 * rk0 + vd1 * rk1 + vd2 * rk2;

    bf16_t* h = H + (size_t)e * 448;
    // Ks flat [0,192): first V_edge (n<32), then Vs (n+32)
    h[3 * lane + 0] = f2bf(ve0);
    h[3 * lane + 1] = f2bf(ve1);
    h[3 * lane + 2] = f2bf(ve2);
    h[96 + 3 * lane + 0] = f2bf(vs0);
    h[96 + 3 * lane + 1] = f2bf(vs1);
    h[96 + 3 * lane + 2] = f2bf(vs2);
    // Qt flat [192,384)
    h[192 + 3 * lane + 0] = f2bf(qe0);
    h[192 + 3 * lane + 1] = f2bf(qe1);
    h[192 + 3 * lane + 2] = f2bf(qe2);
    h[288 + 3 * lane + 0] = f2bf(qs0);
    h[288 + 3 * lane + 1] = f2bf(qs1);
    h[288 + 3 * lane + 2] = f2bf(qs2);
    // quat_st [384,393)
    if (lane < 9) h[384 + lane] = f2bf(R[lane]);
    // RBF of ||trans|| [393,409): centers linspace(0,50,16), std = 50/16
    float nt = sqrtf(t0 * t0 + t1 * t1 + t2 * t2);
    if (lane < 16) {
        float c = (float)lane * (50.0f / 15.0f);
        float z = (nt - c) * (1.0f / 3.125f);
        h[393 + lane] = f2bf(__expf(-z * z));
    }
    // QRK [409,441)
    h[409 + lane] = f2bf(qrk);
    // zero pad [441,448)
    if (lane < 7) h[441 + lane] = f2bf(0.0f);
}

// ---------------------------------------------------------------------------
// Host-side launch
// ---------------------------------------------------------------------------
extern "C" void kernel_launch(void* const* d_in, const int* in_sizes, int n_in,
                              void* d_out, int out_size, void* d_ws, size_t ws_size,
                              hipStream_t stream) {
    const float* h_V   = (const float*)d_in[0];
    const float* h_E   = (const float*)d_in[1];
    const float* rot   = (const float*)d_in[2];
    const float* trans = (const float*)d_in[3];
    const int*   eidx  = (const int*)d_in[4];
    const float* W_va  = (const float*)d_in[5];
    const float* b_va  = (const float*)d_in[6];
    const float* W_vd  = (const float*)d_in[7];
    const float* b_vd  = (const float*)d_in[8];
    const float* W1    = (const float*)d_in[9];
    const float* b1    = (const float*)d_in[10];
    const float* g1    = (const float*)d_in[11];
    const float* be1   = (const float*)d_in[12];
    const float* W2    = (const float*)d_in[13];
    const float* b2    = (const float*)d_in[14];
    const float* g2    = (const float*)d_in[15];
    const float* be2   = (const float*)d_in[16];
    const float* W3    = (const float*)d_in[17];
    const float* b3    = (const float*)d_in[18];
    const float* Wm    = (const float*)d_in[19];
    const float* bm    = (const float*)d_in[20];
    float* out = (float*)d_out;

    const int Nn  = in_sizes[0] / 128;   // 10000 nodes
    const int E   = in_sizes[1] / 128;   // 300000 edges
    const int DIN = in_sizes[9] / 128;   // 441

    // ---- workspace carve-up ----
    char* ws = (char*)d_ws;
    size_t off = 0;
    auto salloc = [&](size_t bytes) -> char* {
        char* p = ws + off;
        off = (off + bytes + 255) & ~(size_t)255;
        return p;
    };
    float*  Vloc  = (float*)salloc((size_t)Nn * 96 * 4);
    float*  Vedge = (float*)salloc((size_t)E * 96 * 4);
    bf16_t* H     = (bf16_t*)salloc((size_t)E * 448 * 2);
    float*  Xf    = (float*)salloc((size_t)E * 128 * 4);
    bf16_t* Ab    = (bf16_t*)salloc((size_t)E * 128 * 2);
    bf16_t* Gb    = (bf16_t*)salloc((size_t)E * 128 * 2);
    bf16_t* hVb   = (bf16_t*)salloc((size_t)Nn * 128 * 2);
    bf16_t* hEb   = (bf16_t*)salloc((size_t)E * 128 * 2);
    bf16_t* WvaP  = (bf16_t*)salloc((size_t)6 * 4 * 512 * 2);
    bf16_t* WvdP  = (bf16_t*)salloc((size_t)6 * 4 * 512 * 2);
    bf16_t* W1P   = (bf16_t*)salloc((size_t)8 * 14 * 512 * 2);
    bf16_t* W2P   = (bf16_t*)salloc((size_t)8 * 4 * 512 * 2);
    bf16_t* W3P   = (bf16_t*)salloc((size_t)8 * 4 * 512 * 2);
    bf16_t* WmTP  = (bf16_t*)salloc((size_t)8 * 4 * 512 * 2);
    bf16_t* WmBP  = (bf16_t*)salloc((size_t)8 * 4 * 512 * 2);
    float*  stats = (float*)salloc(256 * 4);  // colsum[128] + colsumsq[128]
    float*  scsh  = (float*)salloc(256 * 4);  // scale[128] + shift[128]
    float* colsum   = stats;
    float* colsumsq = stats + 128;
    float* scale    = scsh;
    float* shift    = scsh + 128;

    const int MT = 8;  // M-tiles (of 16 rows) per block
    auto gemm_grid = [&](int M) { return dim3((M + 16 * MT - 1) / (16 * MT)); };
    const dim3 blk(256);

    // ---- casts ----
    {
        size_t c1 = (size_t)Nn * 128, c2 = (size_t)E * 128;
        cast_f32_to_bf16<<<dim3((unsigned)((c1 + 1023) / 1024)), blk, 0, stream>>>(h_V, hVb, c1);
        cast_f32_to_bf16<<<dim3((unsigned)((c2 + 1023) / 1024)), blk, 0, stream>>>(h_E, hEb, c2);
    }
    // ---- weight packing ----
    auto pack = [&](const float* W, int N, int Kv, int KT, int nt, bf16_t* dst) {
        int total = nt * KT * 512;
        pack_weights<<<dim3((total + 255) / 256), blk, 0, stream>>>(W, N, Kv, KT, nt, dst);
    };
    pack(W_va, 96, 128, 4, 6, WvaP);
    pack(W_vd, 96, 128, 4, 6, WvdP);
    pack(W1, 128, DIN, 14, 8, W1P);
    pack(W2, 128, 128, 4, 8, W2P);
    pack(W3, 128, 128, 4, 8, W3P);
    pack(Wm, 128, 128, 4, 8, WmTP);                      // rows 0..127  (h_E part)
    pack(Wm + (size_t)128 * 128, 128, 128, 4, 8, WmBP);  // rows 128..255 (G_e part)

    // ---- projections: V_local = h_V @ W_va + b_va ; V_edge = h_E @ W_vd + b_vd
    gemm_wmma<4, 96, false, false, true, false><<<gemm_grid(Nn), blk, 0, stream>>>(
        hVb, 128, WvaP, b_va, Vloc, nullptr, Nn, 6, nullptr, nullptr, MT);
    gemm_wmma<4, 96, false, false, true, false><<<gemm_grid(E), blk, 0, stream>>>(
        hEb, 128, WvdP, b_vd, Vedge, nullptr, E, 6, nullptr, nullptr, MT);

    // ---- per-edge feature rows ----
    build_H<<<dim3((unsigned)(((size_t)E * 32 + 255) / 256)), blk, 0, stream>>>(
        Vloc, Vedge, rot, trans, eidx, E, H);

    const float invE = 1.0f / (float)E;

    // ---- layer 1: X1 = H @ W1 + b1 (+ column stats) ----
    zero_f32<<<dim3(1), blk, 0, stream>>>(stats, 256);
    gemm_wmma<14, 128, true, false, true, false><<<gemm_grid(E), blk, 0, stream>>>(
        H, 448, W1P, b1, Xf, nullptr, E, 8, colsum, colsumsq, MT);
    bn_finalize<<<dim3(1), dim3(128), 0, stream>>>(colsum, colsumsq, g1, be1, invE, scale, shift);
    {
        size_t c = (size_t)E * 128;
        bn_relu_cast<<<dim3((unsigned)((c + 1023) / 1024)), blk, 0, stream>>>(Xf, scale, shift, Ab, c);
    }

    // ---- layer 2 ----
    zero_f32<<<dim3(1), blk, 0, stream>>>(stats, 256);
    gemm_wmma<4, 128, true, false, true, false><<<gemm_grid(E), blk, 0, stream>>>(
        Ab, 128, W2P, b2, Xf, nullptr, E, 8, colsum, colsumsq, MT);
    bn_finalize<<<dim3(1), dim3(128), 0, stream>>>(colsum, colsumsq, g2, be2, invE, scale, shift);
    {
        size_t c = (size_t)E * 128;
        bn_relu_cast<<<dim3((unsigned)((c + 1023) / 1024)), blk, 0, stream>>>(Xf, scale, shift, Ab, c);
    }

    // ---- layer 3: G_e = X2r @ W3 + b3 (store bf16) ----
    gemm_wmma<4, 128, false, false, false, true><<<gemm_grid(E), blk, 0, stream>>>(
        Ab, 128, W3P, b3, nullptr, Gb, E, 8, nullptr, nullptr, MT);

    // ---- output: out = h_E @ Wm_top + bm ; out += G_e @ Wm_bot ----
    gemm_wmma<4, 128, false, false, true, false><<<gemm_grid(E), blk, 0, stream>>>(
        hEb, 128, WmTP, bm, out, nullptr, E, 8, nullptr, nullptr, MT);
    gemm_wmma<4, 128, false, true, true, false><<<gemm_grid(E), blk, 0, stream>>>(
        Gb, 128, WmBP, nullptr, out, nullptr, E, 8, nullptr, nullptr, MT);

    (void)n_in; (void)out_size; (void)ws_size;
}
